// GCNGAT_Encoder_79224966742229
// MI455X (gfx1250) — matrix operand order repaired and checked by reading, compiled, not run on hardware
//
#include <hip/hip_runtime.h>
#include <hip/hip_bf16.h>

#define NN 20000
#define EE 320000
#define ETOT 340000        // EE + NN self loops
#define NH 4               // heads
#define CH 64              // per-head channels (HID == LAT == 64)
#define HC 256             // NH*CH
#define NEG_SLOPE 0.2f

typedef __attribute__((ext_vector_type(16))) _Float16 v16h;
typedef __attribute__((ext_vector_type(8)))  _Float16 v8h;
typedef __attribute__((ext_vector_type(8)))  float    v8f;

// ---------------------------------------------------------------------------
// WMMA GEMM: C[M x Nc] (f32) = A[M x K] (f16 row-major) @ Bt[Nc x K] (f16,
// i.e. W^T row-major == W column-major). One wave32 per 16x16 C tile.
// Fragment layouts per CDNA5 ISA 7.12.2:
//   A lane(l):  M = l&15, halves [0..7] = K kk+8*(l>>4)..+7, [8..15] = +16
//   B lane(l):  N = l&15, halves [0..7] = K kk+8*(l>>4)..+7, [8..15] = +16
//   C lane(l):  VGPR g -> M = 16*tM + g + 8*(l>>4), N = 16*tN + (l&15)
// Each k-step: 4x global_load_b128 + 1 wmma; K unrolled x2.
// ---------------------------------------------------------------------------
__global__ void wmma_gemm(const _Float16* __restrict__ A, const _Float16* __restrict__ Bt,
                          float* __restrict__ C, int M, int K, int Nc) {
  const int lane = threadIdx.x & 31;
  const int wave = threadIdx.x >> 5;
  const int tile = blockIdx.x * (blockDim.x >> 5) + wave;
  const int tilesN = Nc >> 4;
  const int tM = tile / tilesN;
  const int tN = tile - tM * tilesN;
  if (tM * 16 >= M) return;              // wave-uniform: EXEC stays all-1s
  const int l15 = lane & 15;
  const int half = lane >> 4;
  const int m = tM * 16 + l15;
  const int n = tN * 16 + l15;
  const _Float16* arow = A  + (size_t)m * K + 8 * half;    // 16B aligned
  const _Float16* bcol = Bt + (size_t)n * K + 8 * half;    // 16B aligned
  v8f acc = {};
#pragma unroll 2
  for (int kk = 0; kk < K; kk += 32) {
    v8h a0 = *(const v8h*)(arow + kk);
    v8h a1 = *(const v8h*)(arow + kk + 16);
    v8h b0 = *(const v8h*)(bcol + kk);
    v8h b1 = *(const v8h*)(bcol + kk + 16);
    v16h a = __builtin_shufflevector(a0, a1, 0,1,2,3,4,5,6,7,8,9,10,11,12,13,14,15);
    v16h b = __builtin_shufflevector(b0, b1, 0,1,2,3,4,5,6,7,8,9,10,11,12,13,14,15);
    acc = __builtin_amdgcn_wmma_f32_16x16x32_f16(
        /*neg_a=*/false, a, /*neg_b=*/false, b,
        /*c_mod=*/(short)0, acc, /*reuse_a=*/false, /*reuse_b=*/false);
  }
#pragma unroll
  for (int g = 0; g < 8; ++g)
    C[(size_t)(tM * 16 + g + 8 * half) * Nc + n] = acc[g];
}

// pack W[K x Nc] (f32) -> Wt[Nc x K] (f16); i indexes Wt for coalesced writes
__global__ void pack_wt(const float* __restrict__ W, _Float16* __restrict__ Wt,
                        int K, int Nc) {
  int i = blockIdx.x * blockDim.x + threadIdx.x;
  if (i >= K * Nc) return;
  int n = i / K, k = i - n * K;
  Wt[i] = (_Float16)W[k * Nc + n];
}
__global__ void cvt_f16(const float* __restrict__ src, _Float16* __restrict__ dst, int n) {
  int i = blockIdx.x * blockDim.x + threadIdx.x;
  if (i < n) dst[i] = (_Float16)src[i];
}

// ---------------------------------------------------------------------------
// helpers
// ---------------------------------------------------------------------------
__device__ __forceinline__ void edge_sd(const int* __restrict__ ei, int e, int& s, int& d) {
  if (e < EE) { s = ei[e]; d = ei[EE + e]; }
  else        { s = e - EE; d = e - EE; }   // appended self loops
}
// order-preserving float->uint key for atomicMax (handles negatives)
__device__ __forceinline__ unsigned fkey(float f) {
  unsigned u = __float_as_uint(f);
  return (u & 0x80000000u) ? ~u : (u | 0x80000000u);
}
__device__ __forceinline__ float fdekey(unsigned k) {
  unsigned u = (k & 0x80000000u) ? (k ^ 0x80000000u) : ~k;
  return __uint_as_float(u);
}
#define KEY_NEG_INF 0x007FFFFFu   // fkey(-inf)

__global__ void fill_f32(float* p, float v, int n) {
  int i = blockIdx.x * blockDim.x + threadIdx.x;
  if (i < n) p[i] = v;
}
__global__ void fill_u32(unsigned* p, unsigned v, int n) {
  int i = blockIdx.x * blockDim.x + threadIdx.x;
  if (i < n) p[i] = v;
}

// ---------------------------------------------------------------------------
// GCN: degree count, rsqrt, normalized aggregation
// ---------------------------------------------------------------------------
__global__ void deg_count(const int* __restrict__ ei, float* __restrict__ deg) {
  int e = blockIdx.x * blockDim.x + threadIdx.x;
  if (e >= ETOT) return;
  int s, d; edge_sd(ei, e, s, d);
  atomicAdd(&deg[d], 1.0f);
}
__global__ void deg_to_rsqrt(float* __restrict__ deg) {
  int i = blockIdx.x * blockDim.x + threadIdx.x;
  if (i >= NN) return;
  float v = deg[i];
  deg[i] = (v > 0.0f) ? rsqrtf(v) : 0.0f;
}
__global__ void gcn_agg(const float* __restrict__ xw, const float* __restrict__ dis,
                        const int* __restrict__ ei, float* __restrict__ agg) {
  int i = blockIdx.x * blockDim.x + threadIdx.x;   // ETOT*64 threads
  if (i >= ETOT * CH) return;
  int e = i >> 6, c = i & 63;
  int s, d; edge_sd(ei, e, s, d);
  float nrm = dis[s] * dis[d];
  atomicAdd(&agg[d * CH + c], nrm * xw[s * CH + c]);
}

// ---------------------------------------------------------------------------
// GAT pieces
// ---------------------------------------------------------------------------
__global__ void attn_proj(const float* __restrict__ xw, const float* __restrict__ asrc,
                          const float* __restrict__ adst,
                          float* __restrict__ als, float* __restrict__ ald) {
  int i = blockIdx.x * blockDim.x + threadIdx.x;   // NN*NH
  if (i >= NN * NH) return;
  int h = i & (NH - 1);
  const float* v  = xw + (size_t)i * CH;
  const float* a1 = asrc + h * CH;
  const float* a2 = adst + h * CH;
  float s1 = 0.f, s2 = 0.f;
#pragma unroll 8
  for (int c = 0; c < CH; ++c) { float x = v[c]; s1 += x * a1[c]; s2 += x * a2[c]; }
  als[i] = s1; ald[i] = s2;
}

__global__ void logits_max(const float* __restrict__ als, const float* __restrict__ ald,
                           const int* __restrict__ ei, float* __restrict__ alpha,
                           unsigned* __restrict__ mkey) {
  int i = blockIdx.x * blockDim.x + threadIdx.x;   // ETOT*NH
  if (i >= ETOT * NH) return;
  int e = i >> 2, h = i & 3;
  int s, d; edge_sd(ei, e, s, d);
  float l = als[s * NH + h] + ald[d * NH + h];
  l = (l > 0.f) ? l : NEG_SLOPE * l;               // leaky_relu
  alpha[i] = l;                                    // stash logit in output slot
  atomicMax(&mkey[d * NH + h], fkey(l));
}

__global__ void exp_sum(const unsigned* __restrict__ mkey, const int* __restrict__ ei,
                        float* __restrict__ alpha, float* __restrict__ ssum) {
  int i = blockIdx.x * blockDim.x + threadIdx.x;
  if (i >= ETOT * NH) return;
  int e = i >> 2, h = i & 3;
  int s, d; edge_sd(ei, e, s, d);
  float m = fdekey(mkey[d * NH + h]);
  float ex = expf(alpha[i] - m);
  alpha[i] = ex;
  atomicAdd(&ssum[d * NH + h], ex);
}

__global__ void norm_agg(float* __restrict__ alpha, const float* __restrict__ ssum,
                         const float* __restrict__ xw, const int* __restrict__ ei,
                         float* __restrict__ agg) {
  int i = blockIdx.x * blockDim.x + threadIdx.x;
  if (i >= ETOT * NH) return;
  int e = i >> 2, h = i & 3;
  int s, d; edge_sd(ei, e, s, d);
  float al = alpha[i] / (ssum[d * NH + h] + 1e-16f);
  alpha[i] = al;                                   // final softmax output
  const float* src = xw + (size_t)(s * NH + h) * CH;
  float* dst = agg + (size_t)(d * NH + h) * CH;
#pragma unroll 8
  for (int c = 0; c < CH; ++c) atomicAdd(&dst[c], al * src[c]);
}

// finalize variants: h activations are only consumed as GEMM A-operands,
// so emit them directly in f16.
__global__ void bias_relu_f16(const float* __restrict__ agg, const float* __restrict__ b,
                              _Float16* __restrict__ out, int total, int C) {
  int i = blockIdx.x * blockDim.x + threadIdx.x;
  if (i >= total) return;
  float v = agg[i] + b[i % C];
  out[i] = (_Float16)fmaxf(v, 0.f);
}
__global__ void headmean_bias_f16(const float* __restrict__ agg, const float* __restrict__ b,
                                  _Float16* __restrict__ out) {
  int i = blockIdx.x * blockDim.x + threadIdx.x;   // NN*CH
  if (i >= NN * CH) return;
  int n = i >> 6, c = i & 63;
  const float* r = agg + (size_t)n * HC + c;
  float v = 0.25f * (r[0] + r[CH] + r[2 * CH] + r[3 * CH]) + b[c];
  out[i] = (_Float16)fmaxf(v, 0.f);                // gat2: relu
}
__global__ void headmean_bias_f32(const float* __restrict__ agg, const float* __restrict__ b,
                                  float* __restrict__ out) {
  int i = blockIdx.x * blockDim.x + threadIdx.x;   // NN*CH
  if (i >= NN * CH) return;
  int n = i >> 6, c = i & 63;
  const float* r = agg + (size_t)n * HC + c;
  out[i] = 0.25f * (r[0] + r[CH] + r[2 * CH] + r[3 * CH]) + b[c];  // no relu
}

// ---------------------------------------------------------------------------
// host-side layer driver
// ---------------------------------------------------------------------------
static void gat_layer(const _Float16* af16, int K, const float* W, const float* asrc,
                      const float* adst, const float* b,
                      int mode /*0 concat+relu->f16, 1 mean+relu->f16, 2 mean->f32*/,
                      float* alpha, float* xw, float* agg,
                      _Float16* hout16, float* hout32, _Float16* wt16,
                      float* als, float* ald, float* ssum, unsigned* mkey,
                      const int* ei, hipStream_t stream) {
  const int TPB = 256;
  pack_wt<<<(K * HC + TPB - 1) / TPB, TPB, 0, stream>>>(W, wt16, K, HC);
  wmma_gemm<<<(NN / 16) * (HC / 16) / 8, TPB, 0, stream>>>(af16, wt16, xw, NN, K, HC);
  attn_proj<<<(NN * NH + TPB - 1) / TPB, TPB, 0, stream>>>(xw, asrc, adst, als, ald);
  fill_u32<<<(NN * NH + TPB - 1) / TPB, TPB, 0, stream>>>(mkey, KEY_NEG_INF, NN * NH);
  fill_f32<<<(NN * NH + TPB - 1) / TPB, TPB, 0, stream>>>(ssum, 0.f, NN * NH);
  fill_f32<<<(NN * HC + TPB - 1) / TPB, TPB, 0, stream>>>(agg, 0.f, NN * HC);
  logits_max<<<(ETOT * NH + TPB - 1) / TPB, TPB, 0, stream>>>(als, ald, ei, alpha, mkey);
  exp_sum<<<(ETOT * NH + TPB - 1) / TPB, TPB, 0, stream>>>(mkey, ei, alpha, ssum);
  norm_agg<<<(ETOT * NH + TPB - 1) / TPB, TPB, 0, stream>>>(alpha, ssum, xw, ei, agg);
  if (mode == 0)
    bias_relu_f16<<<(NN * HC + TPB - 1) / TPB, TPB, 0, stream>>>(agg, b, hout16, NN * HC, HC);
  else if (mode == 1)
    headmean_bias_f16<<<(NN * CH + TPB - 1) / TPB, TPB, 0, stream>>>(agg, b, hout16);
  else
    headmean_bias_f32<<<(NN * CH + TPB - 1) / TPB, TPB, 0, stream>>>(agg, b, hout32);
}

extern "C" void kernel_launch(void* const* d_in, const int* in_sizes, int n_in,
                              void* d_out, int out_size, void* d_ws, size_t ws_size,
                              hipStream_t stream) {
  (void)in_sizes; (void)n_in; (void)out_size; (void)ws_size;
  const float* x      = (const float*)d_in[0];
  const int*   ei     = (const int*)d_in[1];
  const float* gcnW   = (const float*)d_in[2];
  const float* gcnB   = (const float*)d_in[3];
  const float* g1W    = (const float*)d_in[4];
  const float* g1as   = (const float*)d_in[5];
  const float* g1ad   = (const float*)d_in[6];
  const float* g1b    = (const float*)d_in[7];
  const float* g2W    = (const float*)d_in[8];
  const float* g2as   = (const float*)d_in[9];
  const float* g2ad   = (const float*)d_in[10];
  const float* g2b    = (const float*)d_in[11];
  const float* mW     = (const float*)d_in[12];
  const float* mas    = (const float*)d_in[13];
  const float* mad    = (const float*)d_in[14];
  const float* mb     = (const float*)d_in[15];
  const float* sW     = (const float*)d_in[16];
  const float* sas    = (const float*)d_in[17];
  const float* sad    = (const float*)d_in[18];
  const float* sb     = (const float*)d_in[19];

  // workspace layout (f32 region, then 16B-aligned f16 region)
  float* ws    = (float*)d_ws;
  float* xw    = ws;                              // NN*HC f32
  float* agg   = xw   + (size_t)NN * HC;          // NN*HC f32
  float* deg   = agg  + (size_t)NN * HC;          // NN
  float* als   = deg  + NN;                       // NN*NH
  float* ald   = als  + (size_t)NN * NH;          // NN*NH
  float* ssum  = ald  + (size_t)NN * NH;          // NN*NH
  unsigned* mkey = (unsigned*)(ssum + (size_t)NN * NH); // NN*NH
  _Float16* af16 = (_Float16*)(mkey + (size_t)NN * NH); // NN*HC halves (GEMM A)
  _Float16* wt16 = af16 + (size_t)NN * HC;              // 256*256 halves (packed W^T)

  // output layout (return order): z_mean, z_log_std, a1, a2, a_mean, a_std
  float* out    = (float*)d_out;
  float* z_mean = out;
  float* z_std  = z_mean + (size_t)NN * CH;
  float* a1     = z_std  + (size_t)NN * CH;
  float* a2     = a1 + (size_t)ETOT * NH;
  float* am     = a2 + (size_t)ETOT * NH;
  float* asd    = am + (size_t)ETOT * NH;

  const int TPB = 256;

  // ---- GCN layer: h0 = relu(Dhat^-1/2 A Dhat^-1/2 (xW) + b), h0 -> af16 ----
  cvt_f16<<<(NN * 128 + TPB - 1) / TPB, TPB, 0, stream>>>(x, af16, NN * 128);
  pack_wt<<<(128 * CH + TPB - 1) / TPB, TPB, 0, stream>>>(gcnW, wt16, 128, CH);
  wmma_gemm<<<(NN / 16) * (CH / 16) / 8, TPB, 0, stream>>>(af16, wt16, xw, NN, 128, CH);
  fill_f32<<<(NN + TPB - 1) / TPB, TPB, 0, stream>>>(deg, 0.f, NN);
  deg_count<<<(ETOT + TPB - 1) / TPB, TPB, 0, stream>>>(ei, deg);
  deg_to_rsqrt<<<(NN + TPB - 1) / TPB, TPB, 0, stream>>>(deg);
  fill_f32<<<(NN * CH + TPB - 1) / TPB, TPB, 0, stream>>>(agg, 0.f, NN * CH);
  gcn_agg<<<(ETOT * CH + TPB - 1) / TPB, TPB, 0, stream>>>(xw, deg, ei, agg);
  bias_relu_f16<<<(NN * CH + TPB - 1) / TPB, TPB, 0, stream>>>(agg, gcnB, af16, NN * CH, CH);

  // ---- GAT1 (concat, relu): af16(h0,64) -> af16(h1,256) ----
  gat_layer(af16, CH, g1W, g1as, g1ad, g1b, /*mode=*/0, a1,
            xw, agg, af16, nullptr, wt16, als, ald, ssum, mkey, ei, stream);
  // ---- GAT2 (head-mean, relu): af16(h1,256) -> af16(h2,64) ----
  gat_layer(af16, HC, g2W, g2as, g2ad, g2b, /*mode=*/1, a2,
            xw, agg, af16, nullptr, wt16, als, ald, ssum, mkey, ei, stream);
  // ---- mean head (head-mean, no relu): af16(h2,64) -> z_mean ----
  gat_layer(af16, CH, mW, mas, mad, mb, /*mode=*/2, am,
            xw, agg, nullptr, z_mean, wt16, als, ald, ssum, mkey, ei, stream);
  // ---- std head (head-mean, no relu): af16(h2,64) -> z_log_std ----
  gat_layer(af16, CH, sW, sas, sad, sb, /*mode=*/2, asd,
            xw, agg, nullptr, z_std, wt16, als, ald, ssum, mkey, ei, stream);
}
